// Encoder_18442589569879
// MI455X (gfx1250) — compile-verified, matrix operand order checked by reading, not run on hardware
//
#include <hip/hip_runtime.h>

// GNN: 3 layers of { agg = segment_sum(z[src], dst); h = z + agg;
//                    h = relu(h@w1+b1)@w2 + b2; z = relu(h) }
// N=10000, E=320000, D_IN=128, D_H=256.

#define D_HID 256

typedef __attribute__((ext_vector_type(16))) __bf16 v16bf;
typedef __attribute__((ext_vector_type(8)))  float  v8f;

// agg = z  (float4 copy; replaces zeroing + fused add in GEMM)
__global__ void __launch_bounds__(256) copy4_kernel(float4* __restrict__ dst,
                                                    const float4* __restrict__ src, int n4) {
  int i = blockIdx.x * blockDim.x + threadIdx.x;
  if (i < n4) dst[i] = src[i];
}

// agg[dst[e]*D + c..c+3] += z[src[e]*D + c..c+3]
// One float4 per thread: one coalesced 16B gather + 4 hardware f32 atomics,
// index loads amortized 4x. All traffic L2-resident (whole z/agg ~10MB).
template <int DSHIFT>  // D = 1<<DSHIFT floats per row
__global__ void __launch_bounds__(256) scatter_add_kernel(
    const int* __restrict__ src, const int* __restrict__ dst,
    const float* __restrict__ z, float* __restrict__ agg, int E) {
  const int D = 1 << DSHIFT;
  const int G = D >> 2;                 // float4 groups per row
  int total = E << (DSHIFT - 2);
  int gid = blockIdx.x * blockDim.x + threadIdx.x;
  if (gid >= total) return;
  int e = gid >> (DSHIFT - 2);
  int c = (gid & (G - 1)) << 2;
  int s = src[e];
  int d = dst[e];
  float4 v = *(const float4*)(z + s * D + c);
  float* ap = agg + d * D + c;
  unsafeAtomicAdd(ap + 0, v.x);
  unsafeAtomicAdd(ap + 1, v.y);
  unsafeAtomicAdd(ap + 2, v.z);
  unsafeAtomicAdd(ap + 3, v.w);
}

__device__ __forceinline__ unsigned int pack_bf16x2(float lo, float hi) {
  unsigned short ulo = __builtin_bit_cast(unsigned short, (__bf16)lo);
  unsigned short uhi = __builtin_bit_cast(unsigned short, (__bf16)hi);
  return (unsigned int)ulo | ((unsigned int)uhi << 16);
}

// out[N x 256] = relu( A [N x K] @ W [K x 256] + bias )
// Block: 256 threads = 8 waves. Block tile: 128 rows x 128 cols (grid.y = 2).
// Each wave: 16 rows x 128 cols = 8 WMMA C tiles per 32-deep K step.
// __launch_bounds__(256,1): 2 waves/SIMD -> no spills for 64-VGPR acc set.
// LDS tiles stored PRE-SWIZZLED into per-lane WMMA fragment order:
//   frag element j of lane (half,m): k = 16*(p>=4) + 8*half + 2*(p&3) + (j&1), p=j>>1
//   => writer inverts: half=k[3], p=4*k[4]+k[2:1], off=k[0]; k-pairs pack to one dword.
// Reader: one 32-byte contiguous per-lane load (2x ds_load_b128) per fragment.
template <int K>
__global__ void __launch_bounds__(256, 1) gemm_bias_relu_kernel(
    const float* __restrict__ A,
    const float* __restrict__ W, const float* __restrict__ bias,
    float* __restrict__ out, int nrows) {
  // [tile][lane][8 dwords] : A has 8 row-tiles, B has 8 col-tiles
  __shared__ __align__(16) unsigned int sA[8 * 32 * 8];  // 8 KB
  __shared__ __align__(16) unsigned int sB[8 * 32 * 8];  // 8 KB

  const int tid  = threadIdx.x;
  const int lane = tid & 31;
  const int wv   = tid >> 5;        // wave id 0..7 -> row tile
  const int m    = lane & 15;
  const int half = lane >> 4;
  const int rowBase = blockIdx.x * 128;
  const int colBase = blockIdx.y * 128;
  const int rowMax = nrows - 1;

  v8f acc[8];
#pragma unroll
  for (int t = 0; t < 8; ++t) acc[t] = (v8f){0, 0, 0, 0, 0, 0, 0, 0};

  for (int kk = 0; kk < K; kk += 32) {
    // ---- A tile: 128 rows x 16 k-pairs. Clamped (branch-free) coalesced float2
    //      loads, packed b32 LDS stores. Garbage rows only feed guarded stores.
#pragma unroll
    for (int i = 0; i < 8; ++i) {
      int idx = tid + i * 256;       // 0..2047
      int r = idx >> 4;              // 0..127
      int q = idx & 15;              // k-pair index, k = 2q
      int grow = rowBase + r;
      grow = grow > rowMax ? rowMax : grow;
      const float2* ap = (const float2*)(A + grow * K + kk);
      float2 v = ap[q];
      int h = (q >> 2) & 1;                       // k[3]
      int p = ((q >> 3) & 1) * 4 + (q & 3);       // 4*k[4] + k[2:1]
      int lw = h * 16 + (r & 15);
      sA[(r >> 4) * 256 + lw * 8 + p] = pack_bf16x2(v.x, v.y);
    }
    // ---- B tile: 16 k-pairs x 128 cols from W (K x 256 row-major).
#pragma unroll
    for (int i = 0; i < 8; ++i) {
      int idx = tid + i * 256;       // 0..2047
      int q = idx >> 7;              // k-pair, k = 2q
      int c = idx & 127;
      const float* wp = W + (kk + 2 * q) * D_HID + colBase + c;
      float v0 = wp[0];
      float v1 = wp[D_HID];
      int h = (q >> 2) & 1;
      int p = ((q >> 3) & 1) * 4 + (q & 3);
      int lw = h * 16 + (c & 15);
      sB[(c >> 4) * 256 + lw * 8 + p] = pack_bf16x2(v0, v1);
    }
    __syncthreads();

    v16bf a = *(const v16bf*)(sA + wv * 256 + lane * 8);
#pragma unroll
    for (int t = 0; t < 8; ++t) {
      v16bf b = *(const v16bf*)(sB + t * 256 + lane * 8);
      acc[t] = __builtin_amdgcn_wmma_f32_16x16x32_bf16(
          false, a, false, b, (short)0, acc[t], false, false);
    }
    __syncthreads();
  }

  // Epilogue: D layout — VGPR r holds M = r + 8*half, N = lane%16.
#pragma unroll
  for (int t = 0; t < 8; ++t) {
    int gcol = colBase + t * 16 + m;
    float bv = bias[gcol];
#pragma unroll
    for (int r = 0; r < 8; ++r) {
      int grow = rowBase + wv * 16 + r + half * 8;
      if (grow < nrows) {
        float v = acc[t][r] + bv;
        v = v > 0.0f ? v : 0.0f;  // ReLU (both GEMMs end in ReLU here)
        out[grow * D_HID + gcol] = v;
      }
    }
  }
}

extern "C" void kernel_launch(void* const* d_in, const int* in_sizes, int n_in,
                              void* d_out, int out_size, void* d_ws, size_t ws_size,
                              hipStream_t stream) {
  const float* x  = (const float*)d_in[0];
  const int*   ei = (const int*)d_in[1];
  const int N = in_sizes[0] / 128;     // 10000
  const int E = in_sizes[1] / 2;       // 320000

  const float* w1[3] = {(const float*)d_in[2], (const float*)d_in[6],  (const float*)d_in[10]};
  const float* b1[3] = {(const float*)d_in[3], (const float*)d_in[7],  (const float*)d_in[11]};
  const float* w2[3] = {(const float*)d_in[4], (const float*)d_in[8],  (const float*)d_in[12]};
  const float* b2[3] = {(const float*)d_in[5], (const float*)d_in[9],  (const float*)d_in[13]};

  float* agg  = (float*)d_ws;                      // N*256 f32
  float* tbuf = agg  + (size_t)N * D_HID;          // N*256 f32
  float* zws  = tbuf + (size_t)N * D_HID;          // N*256 f32
  float* dout = (float*)d_out;

  const int* src = ei;
  const int* dst = ei + E;

  dim3 gemmGrid((N + 127) / 128, D_HID / 128);

  for (int l = 0; l < 3; ++l) {
    const float* zin = (l == 0) ? x : zws;
    const int D = (l == 0) ? 128 : 256;

    // agg = z  (so scatter accumulates h = z + sum directly)
    int n4 = (N * D) / 4;
    copy4_kernel<<<(n4 + 255) / 256, 256, 0, stream>>>((float4*)agg, (const float4*)zin, n4);

    // agg[dst] += z[src]  (one float4 per thread)
    int total4 = (E * D) / 4;
    int sblocks = (total4 + 255) / 256;
    if (l == 0)
      scatter_add_kernel<7><<<sblocks, 256, 0, stream>>>(src, dst, zin, agg, E);
    else
      scatter_add_kernel<8><<<sblocks, 256, 0, stream>>>(src, dst, zin, agg, E);

    // t = relu(agg @ w1 + b1)
    if (l == 0)
      gemm_bias_relu_kernel<128><<<gemmGrid, 256, 0, stream>>>(agg, w1[l], b1[l], tbuf, N);
    else
      gemm_bias_relu_kernel<256><<<gemmGrid, 256, 0, stream>>>(agg, w1[l], b1[l], tbuf, N);

    // z = relu(t @ w2 + b2)   (last layer writes d_out directly)
    float* o2 = (l == 2) ? dout : zws;
    gemm_bias_relu_kernel<256><<<gemmGrid, 256, 0, stream>>>(tbuf, w2[l], b2[l], o2, N);
  }
}
